// Model_89197880803466
// MI455X (gfx1250) — matrix operand order repaired and checked by reading, not compiled, not run on hardware
//
#include <hip/hip_runtime.h>

// ---------------- problem constants ----------------
#define TOK     2048          // B*S = 4*512
#define DMODEL  768
#define VOCAB   50264
#define NEDGE   8192
#define NCHUNK  16
#define NTILES  3142          // ceil(VOCAB/16); last tile has 8 valid cols
#define W1TILES 48            // DMODEL/16
#define KSTEPS  24            // DMODEL/32
#define TPC     197           // tiles per chunk (16*197 >= 3142)
#define ASTRIDE 776           // padded bf16 row stride (bank-conflict-free)

typedef __attribute__((ext_vector_type(16))) __bf16 v16bf;
typedef __attribute__((ext_vector_type(8)))  float  v8f;

static __device__ __forceinline__ unsigned short f2bf(float f) {
    union { float f; unsigned u; } x; x.f = f;
    unsigned r = x.u + 0x7FFFu + ((x.u >> 16) & 1u);   // RNE
    return (unsigned short)(r >> 16);
}
static __device__ __forceinline__ float bf2f(unsigned short h) {
    union { unsigned u; float f; } x; x.u = ((unsigned)h) << 16;
    return x.f;
}

// A fragment (16-bit A 16x32): lane<16 -> row=lane,    K = {k0..k0+15}
//                              lane>=16 -> row=lane-16, K = {k0+16..k0+31}
static __device__ __forceinline__ v16bf load_a_frag(const unsigned short* Ash,
                                                    int lane, int k0) {
    int row = lane & 15;
    int hi  = lane >> 4;
    union { v16bf v; uint4 q[2]; } a;
    const unsigned short* p = Ash + row * ASTRIDE + k0 + hi * 16;
    a.q[0] = *(const uint4*)(p);
    a.q[1] = *(const uint4*)(p + 8);
    return a.v;
}

// B fragment from PRE-SWIZZLED bf16: per (tile, kstep) a contiguous 1KB block,
// lane l's 16 bf16 at element offset l*16 -> two coalesced b128 loads.
static __device__ __forceinline__ v16bf load_b_pre(const unsigned short* p) {
    union { v16bf v; uint4 q[2]; } b;
    b.q[0] = *(const uint4*)(p);
    b.q[1] = *(const uint4*)(p + 8);
    return b.v;
}

// Fallback: B fragment built from fp32 rows with in-register conversion.
static __device__ __forceinline__ v16bf load_b_frag_f32(const float* rowK) {
    union { v16bf v; unsigned short u[16]; } b;
    const float4* p = (const float4*)rowK;
#pragma unroll
    for (int i = 0; i < 4; ++i) {
        float4 f = p[i];
        b.u[4*i+0] = f2bf(f.x); b.u[4*i+1] = f2bf(f.y);
        b.u[4*i+2] = f2bf(f.z); b.u[4*i+3] = f2bf(f.w);
    }
    return b.v;
}

// =====================================================================
// Kernel 0: one-time fp32 -> bf16 swizzle into B-fragment layout.
// Block = one (tile, kstep): 512 bf16 out, contiguous. 256 thr x 2 elems.
// =====================================================================
__global__ void __launch_bounds__(256)
k_swizzle(const float* __restrict__ W, unsigned short* __restrict__ out,
          int nrows) {
    const int st  = blockIdx.x;            // global step index = nt*KSTEPS + kk
    const int nt  = st / KSTEPS;
    const int kk  = st % KSTEPS;
    unsigned short* dst = out + (size_t)st * 512;
#pragma unroll
    for (int h = 0; h < 2; ++h) {
        int e    = threadIdx.x + h * 256;  // 0..511
        int lane = e >> 4, i = e & 15;
        int row  = nt * 16 + (lane & 15);
        row      = min(row, nrows - 1);
        int k    = kk * 32 + (lane >> 4) * 16 + i;
        dst[e]   = f2bf(W[(size_t)row * DMODEL + k]);
    }
}

// =====================================================================
// Kernel 1: h = LayerNorm(ReLU(X @ W1^T)) * gamma + beta  -> bf16 ws
// one workgroup per 16-token block, 256 threads = 8 waves,
// dual-N-tile per wave (A fragment shared by 2 WMMAs).
// =====================================================================
template <bool PRE>
__global__ void __launch_bounds__(256)
k_mlm_hidden(const float* __restrict__ X, const float* __restrict__ W1,
             const unsigned short* __restrict__ W1sw,
             const float* __restrict__ gamma, const float* __restrict__ beta,
             unsigned short* __restrict__ hbf) {
    __shared__ unsigned short Ash[16 * ASTRIDE];
    __shared__ unsigned short Hsh[16 * ASTRIDE];
    __shared__ float rsum[16][16], rsq[16][16], rmu[16], rrs[16];

    const int tid  = threadIdx.x;
    const int wv   = tid >> 5;
    const int lane = tid & 31;
    const size_t tokbase = (size_t)blockIdx.x * 16;

    for (int idx = tid; idx < 16 * DMODEL; idx += 256) {
        int r = idx / DMODEL, c = idx % DMODEL;
        Ash[r * ASTRIDE + c] = f2bf(X[(tokbase + r) * DMODEL + c]);
    }
    __syncthreads();

    const int colN = lane & 15;
    const int hi   = lane >> 4;
    for (int nt0 = wv * 2; nt0 < W1TILES; nt0 += 16) {
        const int nt1 = nt0 + 1;                       // always < 48 here
        v8f acc0 = {}, acc1 = {};
        const unsigned short* b0 = W1sw + ((size_t)nt0 * KSTEPS) * 512 + lane * 16;
        const unsigned short* b1 = W1sw + ((size_t)nt1 * KSTEPS) * 512 + lane * 16;
        const int nrow0 = nt0 * 16 + colN;
        const int nrow1 = nt1 * 16 + colN;
#pragma unroll 4
        for (int kk = 0; kk < KSTEPS; ++kk) {
            const int k0 = kk * 32;
            v16bf a  = load_a_frag(Ash, lane, k0);
            v16bf q0 = PRE ? load_b_pre(b0 + (size_t)kk * 512)
                           : load_b_frag_f32(W1 + (size_t)nrow0 * DMODEL + k0 + hi * 16);
            v16bf q1 = PRE ? load_b_pre(b1 + (size_t)kk * 512)
                           : load_b_frag_f32(W1 + (size_t)nrow1 * DMODEL + k0 + hi * 16);
            acc0 = __builtin_amdgcn_wmma_f32_16x16x32_bf16(false, a, false, q0,
                                                           (short)0, acc0, false, false);
            acc1 = __builtin_amdgcn_wmma_f32_16x16x32_bf16(false, a, false, q1,
                                                           (short)0, acc1, false, false);
        }
#pragma unroll
        for (int r = 0; r < 8; ++r) {
            int M = r + hi * 8;
            Hsh[M * ASTRIDE + nt0 * 16 + colN] = f2bf(fmaxf(acc0[r], 0.0f));
            Hsh[M * ASTRIDE + nt1 * 16 + colN] = f2bf(fmaxf(acc1[r], 0.0f));
        }
    }
    __syncthreads();

    {   // LayerNorm stats
        int row = tid >> 4, sub = tid & 15;
        float s = 0.f, s2 = 0.f;
        for (int j = sub; j < DMODEL; j += 16) {
            float v = bf2f(Hsh[row * ASTRIDE + j]);
            s += v; s2 += v * v;
        }
        rsum[row][sub] = s; rsq[row][sub] = s2;
    }
    __syncthreads();
    if ((tid & 15) == 0) {
        int row = tid >> 4;
        float s = 0.f, s2 = 0.f;
        for (int j = 0; j < 16; ++j) { s += rsum[row][j]; s2 += rsq[row][j]; }
        float mu  = s / (float)DMODEL;
        float var = s2 / (float)DMODEL - mu * mu;
        rmu[row] = mu;
        rrs[row] = rsqrtf(var + 1e-5f);
    }
    __syncthreads();
    for (int idx = tid; idx < 16 * DMODEL; idx += 256) {
        int r = idx / DMODEL, c = idx % DMODEL;
        float v = (bf2f(Hsh[r * ASTRIDE + c]) - rmu[r]) * rrs[r] * gamma[c] + beta[c];
        hbf[(tokbase + r) * DMODEL + c] = f2bf(v);
    }
}

// =====================================================================
// Kernel 2: logits = h @ W2^T with fused online log-sum-exp + label pick
// grid = (128 token blocks, 16 vocab chunks), 256 threads = 8 waves,
// dual-N-tile per wave.
// =====================================================================
template <bool PRE>
__global__ void __launch_bounds__(256)
k_vocab_lse(const unsigned short* __restrict__ hbf, const float* __restrict__ W2,
            const unsigned short* __restrict__ W2sw,
            const long long* __restrict__ labels,
            float* __restrict__ pm, float* __restrict__ ps,
            float* __restrict__ glabel) {
    __shared__ unsigned short Ash[16 * ASTRIDE];
    __shared__ float redm[8][16], reds[8][16];

    const int tid  = threadIdx.x;
    const int wv   = tid >> 5;
    const int lane = tid & 31;
    const int hi   = lane >> 4;
    const int colN = lane & 15;
    const int chnk = blockIdx.y;
    const size_t tokbase = (size_t)blockIdx.x * 16;

    {   // A block (bf16) -> LDS, 16B chunks
        const uint4* src = (const uint4*)(hbf + tokbase * DMODEL);
        for (int c = tid; c < 16 * DMODEL / 8; c += 256) {
            int s8i = c * 8;
            int r = s8i / DMODEL, col = s8i % DMODEL;
            *(uint4*)(Ash + r * ASTRIDE + col) = src[c];
        }
    }
    __syncthreads();

    int lbl[8];
#pragma unroll
    for (int r = 0; r < 8; ++r)
        lbl[r] = (int)labels[tokbase + r + hi * 8];

    float m8[8], s8[8];
#pragma unroll
    for (int r = 0; r < 8; ++r) { m8[r] = -1e30f; s8[r] = 0.f; }

    const int tile_begin = chnk * TPC;
    const int tile_end   = min(tile_begin + TPC, NTILES);

    // online-softmax accumulate of one 16x16 logits tile
    auto epi = [&](const v8f& acc, int nt) {
        const int vcol = nt * 16 + colN;
        if (vcol < VOCAB) {
#pragma unroll
            for (int r = 0; r < 8; ++r) {
                float x = acc[r];
                if (lbl[r] == vcol)
                    glabel[tokbase + r + hi * 8] = x;
                float mn = fmaxf(m8[r], x);
                s8[r] = s8[r] * __expf(m8[r] - mn) + __expf(x - mn);
                m8[r] = mn;
            }
        }
    };

    for (int nt0 = tile_begin + wv * 2; nt0 < tile_end; nt0 += 16) {
        const int  nt1  = nt0 + 1;
        const bool has1 = (nt1 < tile_end);            // wave-uniform
        const int  vr0  = min(nt0 * 16 + colN, VOCAB - 1);
        const int  vr1  = min((has1 ? nt1 : nt0) * 16 + colN, VOCAB - 1);
        const unsigned short* b0 = W2sw + ((size_t)nt0 * KSTEPS) * 512 + lane * 16;
        const unsigned short* b1 = W2sw + ((size_t)(has1 ? nt1 : nt0) * KSTEPS) * 512
                                   + lane * 16;

        // prefetch the wave's next tile pair
        int nxt = nt0 + 16;
        if (nxt < tile_end) {
            if (PRE)
                __builtin_prefetch(W2sw + ((size_t)nxt * KSTEPS) * 512 + lane * 16, 0, 0);
            else
                __builtin_prefetch(W2 + (size_t)min(nxt * 16 + colN, VOCAB - 1) * DMODEL
                                      + hi * 16, 0, 0);
        }

        v8f acc0 = {}, acc1 = {};
#pragma unroll 4
        for (int kk = 0; kk < KSTEPS; ++kk) {
            const int k0 = kk * 32;
            v16bf a  = load_a_frag(Ash, lane, k0);
            v16bf q0 = PRE ? load_b_pre(b0 + (size_t)kk * 512)
                           : load_b_frag_f32(W2 + (size_t)vr0 * DMODEL + k0 + hi * 16);
            v16bf q1 = PRE ? load_b_pre(b1 + (size_t)kk * 512)
                           : load_b_frag_f32(W2 + (size_t)vr1 * DMODEL + k0 + hi * 16);
            acc0 = __builtin_amdgcn_wmma_f32_16x16x32_bf16(false, a, false, q0,
                                                           (short)0, acc0, false, false);
            acc1 = __builtin_amdgcn_wmma_f32_16x16x32_bf16(false, a, false, q1,
                                                           (short)0, acc1, false, false);
        }
        epi(acc0, nt0);
        if (has1) epi(acc1, nt1);
    }

    // merge across the 16 lanes of each half-wave (wave32 butterfly)
#pragma unroll
    for (int off = 1; off < 16; off <<= 1) {
#pragma unroll
        for (int r = 0; r < 8; ++r) {
            float om = __shfl_xor(m8[r], off, 32);
            float os = __shfl_xor(s8[r], off, 32);
            float mn = fmaxf(m8[r], om);
            s8[r] = s8[r] * __expf(m8[r] - mn) + os * __expf(om - mn);
            m8[r] = mn;
        }
    }
    if (colN == 0) {
#pragma unroll
        for (int r = 0; r < 8; ++r) {
            redm[wv][hi * 8 + r] = m8[r];
            reds[wv][hi * 8 + r] = s8[r];
        }
    }
    __syncthreads();

    if (tid < 16) {
        float m = -1e30f, s = 0.f;
        for (int w = 0; w < 8; ++w) {
            float om = redm[w][tid], os = reds[w][tid];
            float mn = fmaxf(m, om);
            s = s * __expf(m - mn) + os * __expf(om - mn);
            m = mn;
        }
        size_t t = tokbase + tid;
        pm[t * NCHUNK + chnk] = m;
        ps[t * NCHUNK + chnk] = s;
    }
}

// =====================================================================
// Kernel 3: edge head (tiny): p = sigmoid(<df1 t1, df2 t2>), BCE term
// =====================================================================
__global__ void __launch_bounds__(256)
k_edges(const float* __restrict__ out_df,
        const float* __restrict__ df1, const float* __restrict__ df2,
        const int* __restrict__ eb, const int* __restrict__ ej,
        const int* __restrict__ ek, const float* __restrict__ ylab,
        float* __restrict__ terms) {
    __shared__ float t1[DMODEL], t2[DMODEL], prod[64];
    const int tid = threadIdx.x;
    const int e   = blockIdx.x;

    const int b = eb[e], j = ej[e], k = ek[e];
    const float* p1 = out_df + ((size_t)(b * 512 + j)) * DMODEL;
    const float* p2 = out_df + ((size_t)(b * 512 + k)) * DMODEL;
    for (int i = tid; i < DMODEL; i += 256) { t1[i] = p1[i]; t2[i] = p2[i]; }
    __syncthreads();

    const int f = tid >> 2, part = tid & 3;
    const float* wa = df1 + (size_t)f * DMODEL + part * 192;
    const float* wb = df2 + (size_t)f * DMODEL + part * 192;
    float s1 = 0.f, s2 = 0.f;
    for (int d = 0; d < 192; ++d) {
        s1 += wa[d] * t1[part * 192 + d];
        s2 += wb[d] * t2[part * 192 + d];
    }
    s1 += __shfl_xor(s1, 1, 32); s1 += __shfl_xor(s1, 2, 32);
    s2 += __shfl_xor(s2, 1, 32); s2 += __shfl_xor(s2, 2, 32);
    if (part == 0) prod[f] = s1 * s2;
    __syncthreads();

    if (tid == 0) {
        float z = 0.f;
        for (int i = 0; i < 64; ++i) z += prod[i];
        float p   = 1.0f / (1.0f + __expf(-z));
        float lp  = fmaxf(logf(p), -100.0f);
        float l1p = fmaxf(log1pf(-p), -100.0f);
        float y   = ylab[e];
        terms[e]  = -(y * lp + (1.0f - y) * l1p);
    }
}

// =====================================================================
// Kernel 4: finalize — merge chunk partials, masked-mean NLL, edge mean
// =====================================================================
__global__ void __launch_bounds__(256)
k_finalize(const long long* __restrict__ labels,
           const float* __restrict__ pm, const float* __restrict__ ps,
           const float* __restrict__ glabel, const float* __restrict__ terms,
           float* __restrict__ out) {
    __shared__ float s_nll[256], s_cnt[256], s_df[256];
    const int tid = threadIdx.x;

    float nll = 0.f, cnt = 0.f, df = 0.f;
#pragma unroll
    for (int i = 0; i < TOK / 256; ++i) {
        int t = tid + i * 256;
        float m = -1e30f, s = 0.f;
        for (int c = 0; c < NCHUNK; ++c) {
            float om = pm[(size_t)t * NCHUNK + c], os = ps[(size_t)t * NCHUNK + c];
            float mn = fmaxf(m, om);
            s = s * __expf(m - mn) + os * __expf(om - mn);
            m = mn;
        }
        long long lab = labels[t];
        if (lab != 1) {
            nll += (m + logf(s)) - glabel[t];
            cnt += 1.0f;
        }
    }
#pragma unroll
    for (int i = 0; i < NEDGE / 256; ++i)
        df += terms[tid + i * 256];

    s_nll[tid] = nll; s_cnt[tid] = cnt; s_df[tid] = df;
    __syncthreads();
    for (int st = 128; st > 0; st >>= 1) {
        if (tid < st) {
            s_nll[tid] += s_nll[tid + st];
            s_cnt[tid] += s_cnt[tid + st];
            s_df[tid]  += s_df[tid + st];
        }
        __syncthreads();
    }
    if (tid == 0) {
        float loss = s_nll[0] / fmaxf(s_cnt[0], 1.0f);
        float dfl  = s_df[0] / (float)NEDGE;
        out[0] = loss + 0.1f * dfl;
        out[1] = loss;
        out[2] = dfl;
    }
}

// =====================================================================
extern "C" void kernel_launch(void* const* d_in, const int* in_sizes, int n_in,
                              void* d_out, int out_size, void* d_ws, size_t ws_size,
                              hipStream_t stream) {
    const float*     X      = (const float*)d_in[0];
    const float*     out_df = (const float*)d_in[1];
    const float*     W1     = (const float*)d_in[2];
    const float*     gamma  = (const float*)d_in[3];
    const float*     beta   = (const float*)d_in[4];
    const float*     W2     = (const float*)d_in[5];
    const float*     df1    = (const float*)d_in[6];
    const float*     df2    = (const float*)d_in[7];
    const long long* srcids = (const long long*)d_in[8];
    const int*       eb     = (const int*)d_in[9];
    const int*       ej     = (const int*)d_in[10];
    const int*       ek     = (const int*)d_in[11];
    const float*     ylab   = (const float*)d_in[12];
    float*           out    = (float*)d_out;

    char* ws = (char*)d_ws;
    unsigned short* hbf    = (unsigned short*)(ws);                    // 3,145,728 B
    float*          pm     = (float*)(ws + 3145728);                   //   131,072 B
    float*          ps     = (float*)(ws + 3276800);                   //   131,072 B
    float*          glabel = (float*)(ws + 3407872);                   //     8,192 B
    float*          terms  = (float*)(ws + 3416064);                   //    32,768 B
    unsigned short* W1sw   = (unsigned short*)(ws + 3448832);          // 1,179,648 B
    unsigned short* W2sw   = (unsigned short*)(ws + 4628480);          // 77,217,792 B
    const size_t NEED = 4628480ull + 77217792ull;                      // ~78 MB

    const bool pre = (ws_size >= NEED);

    if (pre) {
        k_swizzle<<<dim3(W1TILES * KSTEPS), dim3(256), 0, stream>>>(W1, W1sw, DMODEL);
        k_swizzle<<<dim3(NTILES * KSTEPS),  dim3(256), 0, stream>>>(W2, W2sw, VOCAB);
        k_mlm_hidden<true><<<dim3(TOK / 16), dim3(256), 0, stream>>>(
            X, W1, W1sw, gamma, beta, hbf);
        k_vocab_lse<true><<<dim3(TOK / 16, NCHUNK), dim3(256), 0, stream>>>(
            hbf, W2, W2sw, srcids, pm, ps, glabel);
    } else {
        k_mlm_hidden<false><<<dim3(TOK / 16), dim3(256), 0, stream>>>(
            X, W1, W1sw, gamma, beta, hbf);
        k_vocab_lse<false><<<dim3(TOK / 16, NCHUNK), dim3(256), 0, stream>>>(
            hbf, W2, W2sw, srcids, pm, ps, glabel);
    }
    k_edges<<<dim3(NEDGE), dim3(256), 0, stream>>>(out_df, df1, df2, eb, ej, ek,
                                                   ylab, terms);
    k_finalize<<<dim3(1), dim3(256), 0, stream>>>(srcids, pm, ps, glabel, terms, out);
}
